// MTLRecurrentGCN_10660108829012
// MI455X (gfx1250) — matrix operand order repaired and checked by reading, not compiled, not run on hardware
//
#include <hip/hip_runtime.h>
#include <hip/hip_bf16.h>
#include <math.h>

// ---------------- problem constants (match reference) ----------------
#define N_NODES 50000
#define NPAD    50016              // multiple of 16
#define MT      (NPAD/16)          // 3126 row tiles
#define F_INN   64
#define HD      32
#define T_STEPS 8
#define E_EDGES 800000
#define EPAIRS  65536
#define KDIM    288                // 3*64 (X cheb) + 3*32 (H cheb)
#define KT      9                  // 288 / 32
#define GOUT    128                // 4 gates * 32
#define LEADC   9

typedef __attribute__((ext_vector_type(16))) _Float16 v16h;
typedef __attribute__((ext_vector_type(8)))  float    v8f;

// ---------------- generic utils ----------------
__global__ void zero_f32(float* __restrict__ p, int n) {
  for (int i = blockIdx.x * blockDim.x + threadIdx.x; i < n; i += gridDim.x * blockDim.x)
    p[i] = 0.f;
}

__device__ __forceinline__ float sigm(float x) { return 1.f / (1.f + expf(-x)); }

// ---------------- per-timestep graph kernels ----------------
__global__ void deg_accum(const int* __restrict__ es, float* __restrict__ deg) {
  int e = blockIdx.x * blockDim.x + threadIdx.x;
  if (e < E_EDGES) atomicAdd(deg + es[e], 1.0f);
}

__global__ void compute_dinv(const float* __restrict__ deg, float* __restrict__ dinv) {
  int i = blockIdx.x * blockDim.x + threadIdx.x;
  if (i >= N_NODES) return;
  float d = deg[i];
  dinv[i] = d > 0.f ? rsqrtf(fmaxf(d, 1e-12f)) : 0.f;
}

// out[dst] += (-dinv[src]*dinv[dst]) * vin[src]   (scatter-add Laplacian apply)
__global__ void lap_scatter(const int* __restrict__ es, const int* __restrict__ ed,
                            const float* __restrict__ dinv,
                            const float* __restrict__ vin, float* __restrict__ vout, int F) {
  int e = blockIdx.x * blockDim.x + threadIdx.x;
  if (e >= E_EDGES) return;
  int s = es[e], d = ed[e];
  float w = -dinv[s] * dinv[d];
  const float4* vs = (const float4*)(vin + (size_t)s * F);
  float* vo = vout + (size_t)d * F;
  int f4 = F >> 2;
  for (int f = 0; f < f4; ++f) {
    float4 v = vs[f];
    atomicAdd(vo + 4*f + 0, w * v.x);
    atomicAdd(vo + 4*f + 1, w * v.y);
    atomicAdd(vo + 4*f + 2, w * v.z);
    atomicAdd(vo + 4*f + 3, w * v.w);
  }
}

// ---------------- operand packing for WMMA ----------------
// A fragment layout (16-bit A 16x32, wave32):
//   lane L: row = L&15 ; element j: K = 8*(L>>4) + (j<8 ? j : j+8)
// A_frag addr = ((mt*KT + kt)*32 + lane)*16 + j  -> GEMM does one contiguous v16h load.
// Cheb-2 fold applied here: order-2 term = 2*lap(lap(v)) - v.
__global__ void build_A(const float* __restrict__ X,  const float* __restrict__ Tx1,
                        const float* __restrict__ Tx2, const float* __restrict__ Hst,
                        const float* __restrict__ Th1, const float* __restrict__ Th2,
                        _Float16* __restrict__ Af) {
  int t = blockIdx.x * blockDim.x + threadIdx.x;
  const int total = MT * KT * 512;
  if (t >= total) return;
  int j    = t & 15;
  int lane = (t >> 4) & 31;
  int kt   = (t >> 9) % KT;
  int mt   = t / (KT * 512);
  int row  = mt * 16 + (lane & 15);
  int kk   = ((lane >> 4) << 3) + (j < 8 ? j : j + 8);
  int k    = kt * 32 + kk;
  float v = 0.f;
  if (row < N_NODES) {
    if      (k < 64)  v = X  [(size_t)row*64 + k];
    else if (k < 128) v = Tx1[(size_t)row*64 + (k-64)];
    else if (k < 192) v = 2.f*Tx2[(size_t)row*64 + (k-128)] - X  [(size_t)row*64 + (k-128)];
    else if (k < 224) v = Hst[(size_t)row*32 + (k-192)];
    else if (k < 256) v = Th1[(size_t)row*32 + (k-224)];
    else              v = 2.f*Th2[(size_t)row*32 + (k-256)] - Hst[(size_t)row*32 + (k-256)];
  }
  Af[t] = (_Float16)v;
}

// B fragment layout (16-bit B 32x16, wave32):
//   lane L: col = L&15 ; element j: K = 8*(L>>4) + (j<8 ? j : j+8)
// W_frag addr = ((kt*8 + nt)*32 + lane)*16 + j
// Column blocks n: [0,32)=i gate, [32,64)=f, [64,96)=c, [96,128)=o.
// Row blocks k: [0,192)=W_x cheb orders 0..2, [192,288)=W_h cheb orders 0..2.
__global__ void pack_weights(const float* __restrict__ Wxi, const float* __restrict__ Wxf,
                             const float* __restrict__ Wxc, const float* __restrict__ Wxo,
                             const float* __restrict__ Whi, const float* __restrict__ Whf,
                             const float* __restrict__ Whc, const float* __restrict__ Who,
                             _Float16* __restrict__ Wf) {
  int t = blockIdx.x * blockDim.x + threadIdx.x;
  if (t >= KDIM * GOUT) return;
  int n = t % GOUT, k = t / GOUT;
  int g = n >> 5, h = n & 31;
  const float* Wx = (g == 0) ? Wxi : (g == 1) ? Wxf : (g == 2) ? Wxc : Wxo;
  const float* Wh = (g == 0) ? Whi : (g == 1) ? Whf : (g == 2) ? Whc : Who;
  float w;
  if (k < 192) { int o = k / 64, fin = k % 64; w = Wx[(o*64 + fin)*32 + h]; }
  else         { int kp = k-192, o = kp >> 5, hin = kp & 31; w = Wh[(o*32 + hin)*32 + h]; }
  int kt = k >> 5, kk = k & 31;
  int lane = (((kk >> 3) & 1) ? 16 : 0) + (n & 15);
  int j = (kk & 7) + ((kk >> 4) << 3);
  int nt = n >> 4;
  Wf[((kt*8 + nt)*32 + lane)*16 + j] = (_Float16)w;
}

__global__ void pack_bias(const float* bxi, const float* bhi, const float* bi,
                          const float* bxf, const float* bhf, const float* bf,
                          const float* bxc, const float* bhc, const float* bc,
                          const float* bxo, const float* bho, const float* bo,
                          float* __restrict__ bias) {
  int n = threadIdx.x;
  if (n >= GOUT) return;
  int g = n >> 5, h = n & 31;
  const float* a = (g==0)?bxi:(g==1)?bxf:(g==2)?bxc:bxo;
  const float* b = (g==0)?bhi:(g==1)?bhf:(g==2)?bhc:bho;
  const float* c = (g==0)?bi :(g==1)?bf :(g==2)?bc :bo;
  bias[n] = a[h] + b[h] + c[h];
}

// ---------------- WMMA GEMM: Z[NPAD,128] = A[NPAD,288] @ W[288,128] ----------------
// one wave per 16x16 output tile; 4 waves (n-tiles) per block; grid (MT, 2)
__global__ void gemm_wmma(const _Float16* __restrict__ Af, const _Float16* __restrict__ Wf,
                          float* __restrict__ Z) {
  int lane = threadIdx.x;
  int mt = blockIdx.x;
  int nt = threadIdx.y + 4 * blockIdx.y;
  v8f acc = {};
  const _Float16* ap = Af + (size_t)mt * KT * 512 + lane * 16;
  const _Float16* bp = Wf + (size_t)nt * 512 + lane * 16;
  for (int kt = 0; kt < KT; ++kt) {
    v16h a = *(const v16h*)ap;  ap += 512;     // next k-tile of this row tile
    v16h b = *(const v16h*)bp;  bp += 4096;    // next k-tile of this col tile
    acc = __builtin_amdgcn_wmma_f32_16x16x32_f16(
        /*neg_a=*/false, a, /*neg_b=*/false, b,
        /*c_mod=*/(short)0, acc, /*reuse_a=*/false, /*reuse_b=*/false);
  }
  // C/D layout: vgpr r -> M = r + 8*(lane>>4); N = lane&15
  int row0 = mt * 16 + 8 * (lane >> 4);
  int col  = nt * 16 + (lane & 15);
  float* zp = Z + (size_t)row0 * GOUT + col;
  #pragma unroll
  for (int r = 0; r < 8; ++r) zp[(size_t)r * GOUT] = acc[r];
}

// ---------------- GConvLSTM gate pointwise update ----------------
__global__ void gate_update(const float* __restrict__ Z, const float* __restrict__ bias,
                            const float* __restrict__ wci, const float* __restrict__ wcf,
                            const float* __restrict__ wco,
                            float* __restrict__ Hst, float* __restrict__ Cst) {
  int t = blockIdx.x * blockDim.x + threadIdx.x;
  if (t >= N_NODES * HD) return;
  int i = t >> 5, h = t & 31;
  const float* zr = Z + (size_t)i * GOUT;
  float c  = Cst[t];
  float I  = sigm(zr[h]      + bias[h]      + wci[h] * c);
  float Fg = sigm(zr[32 + h] + bias[32 + h] + wcf[h] * c);
  float Tg = tanhf(zr[64 + h] + bias[64 + h]);
  float Cn = Fg * c + I * Tg;
  float O  = sigm(zr[96 + h] + bias[96 + h] + wco[h] * Cn);
  Hst[t] = O * tanhf(Cn);
  Cst[t] = Cn;
}

// ---------------- head ----------------
__global__ void node_fc1(const float* __restrict__ Hst, const float* __restrict__ fc1w,
                         const float* __restrict__ fc1b, float* __restrict__ h16) {
  int t = blockIdx.x * blockDim.x + threadIdx.x;
  if (t >= N_NODES * 16) return;
  int i = t >> 4, o = t & 15;
  float a = fc1b[o];
  #pragma unroll
  for (int k = 0; k < 32; ++k) a += fmaxf(Hst[(size_t)i*32 + k], 0.f) * fc1w[k*16 + o];
  h16[t] = fmaxf(a, 0.f);
}

__global__ void pair_head(const float* __restrict__ h16, const int* __restrict__ src,
                          const int* __restrict__ dst,
                          const float* __restrict__ fc2w, const float* __restrict__ fc2b,
                          const float* __restrict__ fc3w, const float* __restrict__ fc3b,
                          const float* __restrict__ binw, const float* __restrict__ binb,
                          const float* __restrict__ leadw, const float* __restrict__ leadb,
                          float* __restrict__ out) {
  __shared__ float s2w[512], s2b[16], s3w[128], s3b[8], sbw[8], slw[72], slb[9], sbb[1];
  int lt = threadIdx.x;
  for (int i = lt; i < 512; i += blockDim.x) s2w[i] = fc2w[i];
  for (int i = lt; i < 128; i += blockDim.x) s3w[i] = fc3w[i];
  for (int i = lt; i < 72;  i += blockDim.x) slw[i] = leadw[i];
  if (lt < 16) s2b[lt] = fc2b[lt];
  if (lt < 8)  { s3b[lt] = fc3b[lt]; sbw[lt] = binw[lt]; }
  if (lt < 9)  slb[lt] = leadb[lt];
  if (lt == 0) sbb[0] = binb[0];
  __syncthreads();
  int e = blockIdx.x * blockDim.x + lt;
  if (e >= EPAIRS) return;
  int si = src[e], di = dst[e];
  float xin[32];
  #pragma unroll
  for (int i = 0; i < 16; ++i) {
    xin[i]      = h16[(size_t)si*16 + i];
    xin[16 + i] = h16[(size_t)di*16 + i];
  }
  float t16[16];
  for (int o = 0; o < 16; ++o) {
    float a = s2b[o];
    for (int k = 0; k < 32; ++k) a += xin[k] * s2w[k*16 + o];
    t16[o] = fmaxf(a, 0.f);
  }
  float t8[8];
  for (int o = 0; o < 8; ++o) {
    float a = s3b[o];
    for (int k = 0; k < 16; ++k) a += t16[k] * s3w[k*8 + o];
    t8[o] = fmaxf(a, 0.f);
  }
  float zb = sbb[0];
  #pragma unroll
  for (int k = 0; k < 8; ++k) zb += t8[k] * sbw[k];
  out[e] = sigm(zb);
  float l[LEADC]; float m = -1e30f;
  for (int c = 0; c < LEADC; ++c) {
    float a = slb[c];
    for (int k = 0; k < 8; ++k) a += t8[k] * slw[k*LEADC + c];
    l[c] = a; m = fmaxf(m, a);
  }
  float ssum = 0.f;
  for (int c = 0; c < LEADC; ++c) ssum += expf(l[c] - m);
  float lse = m + logf(ssum);
  for (int c = 0; c < LEADC; ++c) out[(size_t)EPAIRS + (size_t)e*LEADC + c] = l[c] - lse;
}

// ---------------- launcher ----------------
extern "C" void kernel_launch(void* const* d_in, const int* in_sizes, int n_in,
                              void* d_out, int out_size, void* d_ws, size_t ws_size,
                              hipStream_t stream) {
  (void)in_sizes; (void)n_in; (void)out_size; (void)ws_size;
  const float* X     = (const float*)d_in[0];
  const int*   edges = (const int*)  d_in[1];   // [T,2,E]
  const int*   src   = (const int*)  d_in[2];
  const int*   dst   = (const int*)  d_in[3];
  // params, dict insertion order
  const float* Wx[4], *bx[4], *Wh[4], *bh[4], *bg[4];
  for (int g = 0; g < 4; ++g) {
    Wx[g] = (const float*)d_in[4 + 5*g + 0];
    bx[g] = (const float*)d_in[4 + 5*g + 1];
    Wh[g] = (const float*)d_in[4 + 5*g + 2];
    bh[g] = (const float*)d_in[4 + 5*g + 3];
    bg[g] = (const float*)d_in[4 + 5*g + 4];
  }
  const float* wci  = (const float*)d_in[24];
  const float* wcf  = (const float*)d_in[25];
  const float* wco  = (const float*)d_in[26];
  const float* fc1w = (const float*)d_in[27];
  const float* fc1b = (const float*)d_in[28];
  const float* fc2w = (const float*)d_in[29];
  const float* fc2b = (const float*)d_in[30];
  const float* fc3w = (const float*)d_in[31];
  const float* fc3b = (const float*)d_in[32];
  const float* binw = (const float*)d_in[33];
  const float* binb = (const float*)d_in[34];
  const float* leadw = (const float*)d_in[35];
  const float* leadb = (const float*)d_in[36];

  // workspace carve-out (256B aligned)
  char* base = (char*)d_ws;
  size_t off = 0;
  auto carve = [&](size_t bytes) -> char* {
    char* p = base + off;
    off += (bytes + 255) & ~(size_t)255;
    return p;
  };
  float*    deg  = (float*)   carve((size_t)N_NODES * 4);
  float*    dinv = (float*)   carve((size_t)N_NODES * 4);
  float*    Hst  = (float*)   carve((size_t)N_NODES * HD * 4);
  float*    Cst  = (float*)   carve((size_t)N_NODES * HD * 4);
  float*    Tx1  = (float*)   carve((size_t)N_NODES * F_INN * 4);
  float*    Tx2  = (float*)   carve((size_t)N_NODES * F_INN * 4);
  float*    Th1  = (float*)   carve((size_t)N_NODES * HD * 4);
  float*    Th2  = (float*)   carve((size_t)N_NODES * HD * 4);
  _Float16* Af   = (_Float16*)carve((size_t)NPAD * KDIM * 2);
  _Float16* Wf   = (_Float16*)carve((size_t)KT * 8 * 32 * 16 * 2);
  float*    bias = (float*)   carve((size_t)GOUT * 4);
  float*    Z    = (float*)   carve((size_t)NPAD * GOUT * 4);
  float*    h16  = (float*)   carve((size_t)N_NODES * 16 * 4);

  const int ZB = 256, ZG = 2048;   // zero kernel config (grid-stride)
  const int EB = (E_EDGES + 255) / 256;
  const int NB = (N_NODES + 255) / 256;

  // one-time packs + state init
  pack_weights<<<(KDIM*GOUT + 255)/256, 256, 0, stream>>>(
      Wx[0], Wx[1], Wx[2], Wx[3], Wh[0], Wh[1], Wh[2], Wh[3], Wf);
  pack_bias<<<1, 128, 0, stream>>>(bx[0], bh[0], bg[0], bx[1], bh[1], bg[1],
                                   bx[2], bh[2], bg[2], bx[3], bh[3], bg[3], bias);
  zero_f32<<<ZG, ZB, 0, stream>>>(Hst, N_NODES * HD);
  zero_f32<<<ZG, ZB, 0, stream>>>(Cst, N_NODES * HD);

  for (int t = 0; t < T_STEPS; ++t) {
    const int* es = edges + (size_t)t * 2 * E_EDGES;
    const int* ed = es + E_EDGES;

    zero_f32<<<ZG, ZB, 0, stream>>>(deg, N_NODES);
    deg_accum<<<EB, 256, 0, stream>>>(es, deg);
    compute_dinv<<<NB, 256, 0, stream>>>(deg, dinv);

    zero_f32<<<ZG, ZB, 0, stream>>>(Tx1, N_NODES * F_INN);
    zero_f32<<<ZG, ZB, 0, stream>>>(Tx2, N_NODES * F_INN);
    zero_f32<<<ZG, ZB, 0, stream>>>(Th1, N_NODES * HD);
    zero_f32<<<ZG, ZB, 0, stream>>>(Th2, N_NODES * HD);

    lap_scatter<<<EB, 256, 0, stream>>>(es, ed, dinv, X,   Tx1, F_INN);
    lap_scatter<<<EB, 256, 0, stream>>>(es, ed, dinv, Tx1, Tx2, F_INN);
    lap_scatter<<<EB, 256, 0, stream>>>(es, ed, dinv, Hst, Th1, HD);
    lap_scatter<<<EB, 256, 0, stream>>>(es, ed, dinv, Th1, Th2, HD);

    build_A<<<(MT*KT*512 + 255)/256, 256, 0, stream>>>(X, Tx1, Tx2, Hst, Th1, Th2, Af);

    dim3 gblk(32, 4), ggrd(MT, 2);
    gemm_wmma<<<ggrd, gblk, 0, stream>>>(Af, Wf, Z);

    gate_update<<<(N_NODES*HD + 255)/256, 256, 0, stream>>>(Z, bias, wci, wcf, wco, Hst, Cst);
  }

  node_fc1<<<(N_NODES*16 + 255)/256, 256, 0, stream>>>(Hst, fc1w, fc1b, h16);
  pair_head<<<EPAIRS/256, 256, 0, stream>>>(h16, src, dst, fc2w, fc2b, fc3w, fc3b,
                                            binw, binb, leadw, leadb, (float*)d_out);
}
